// SVC_63737314673237
// MI455X (gfx1250) — matrix operand order repaired and checked by reading, not compiled, not run on hardware
//
#include <hip/hip_runtime.h>

// SVC one-vs-one RBF classifier, fused WMMA + TDM implementation for gfx1250.
//
// Pipeline:
//   prep_x   : x[8192,256] f32 -> bf16, gxn[n] = GAMMA*||x_n||^2
//   prep_sv  : sv[5000,256] -> bf16 padded to [5120,256] (512 rows/class),
//              gsvn[p] = GAMMA*||sv_p||^2 (0 on pad rows)
//   prep_a   : a[9,5000] -> bf16 [16,5120] (r padded to 16, cols padded to 512/class)
//   svc_main : per (row-block of 128, class c): sv chunk (32x256 bf16, 16KB) staged
//              into LDS once per block via tensor_load_to_lds (double-buffered,
//              overlapped with compute); k = exp(2g*x.sv - gxn - gsvn) via
//              v_wmma_f32_16x16x32_bf16; T[n,r,c] += a*k via a second WMMA.
//   svc_vote : 45 pairwise decisions -> votes -> argmax -> int32 class ids.

typedef __attribute__((ext_vector_type(16))) __bf16 v16bf;
typedef __attribute__((ext_vector_type(8)))  __bf16 v8bf;
typedef __attribute__((ext_vector_type(8)))  float  v8f;
typedef __attribute__((ext_vector_type(4)))  unsigned int v4u;
typedef __attribute__((ext_vector_type(8)))  int    v8i;
typedef __attribute__((ext_vector_type(4)))  int    v4i;

#define SVC_GAMMA 0.01f
#define NCLS 10
#define NPAD 512            // padded SVs per class (multiple of 32)
#define SVPAD (NCLS * NPAD) // 5120
#define FEAT 256
#define ROWB 544            // LDS row stride bytes: 512 + 32 pad (17*32, keeps 32B align)

// ---------------------------------------------------------------- prep kernels
__global__ void svc_prep_x(const float* __restrict__ x, __bf16* __restrict__ xb,
                           float* __restrict__ gxn, int N) {
    int n = blockIdx.x * blockDim.x + threadIdx.x;
    if (n >= N) return;
    const float* r = x + (size_t)n * FEAT;
    __bf16* o = xb + (size_t)n * FEAT;
    float s = 0.f;
#pragma unroll 8
    for (int j = 0; j < FEAT; ++j) { float v = r[j]; s += v * v; o[j] = (__bf16)v; }
    gxn[n] = SVC_GAMMA * s;
}

__global__ void svc_prep_sv(const float* __restrict__ sv, __bf16* __restrict__ svb,
                            float* __restrict__ gsvn) {
    int p = blockIdx.x * blockDim.x + threadIdx.x;
    if (p >= SVPAD) return;
    int c = p >> 9, q = p & (NPAD - 1);
    __bf16* o = svb + (size_t)p * FEAT;
    if (q < 500) {
        const float* r = sv + (size_t)(c * 500 + q) * FEAT;
        float s = 0.f;
#pragma unroll 8
        for (int j = 0; j < FEAT; ++j) { float v = r[j]; s += v * v; o[j] = (__bf16)v; }
        gsvn[p] = SVC_GAMMA * s;
    } else {
#pragma unroll 8
        for (int j = 0; j < FEAT; ++j) o[j] = (__bf16)0.f;
        gsvn[p] = 0.f;
    }
}

__global__ void svc_prep_a(const float* __restrict__ a, __bf16* __restrict__ ab) {
    int idx = blockIdx.x * blockDim.x + threadIdx.x;
    if (idx >= 16 * SVPAD) return;
    int r = idx / SVPAD, p = idx % SVPAD;
    int c = p >> 9, q = p & (NPAD - 1);
    float v = (r < NCLS - 1 && q < 500) ? a[(size_t)r * 5000 + c * 500 + q] : 0.f;
    ab[idx] = (__bf16)v;
}

// -------------------------------------------------- TDM: sv chunk -> LDS (async)
// 2D tensor [5120 rows x 256 feats] bf16; tile = 32 rows x 256 feats starting at
// row sbase. LDS destination padded: +8 DWORDs after every 128 DWORDs (one row)
// -> 544B row stride (bank spread, 32B-aligned rows).
__device__ __forceinline__ void svc_tdm_issue(const __bf16* svb, int sbase, unsigned lds_off) {
    unsigned long long ga = (unsigned long long)(uintptr_t)(svb + (size_t)sbase * FEAT);
    v4u g0;
    g0[0] = 1u;                                            // count=1, user descriptor
    g0[1] = lds_off;                                       // lds_addr (bytes)
    g0[2] = (unsigned)(ga & 0xFFFFFFFFu);                  // global_addr[31:0]
    g0[3] = (unsigned)((ga >> 32) & 0x01FFFFFFu) | (2u << 30); // addr[56:32] | type=2
    v8i g1;
    g1[0] = (int)0x0F910000;   // data_size=1(2B) | pad_enable | pad_interval=6(128dw) | pad_amount=7(8dw)
    g1[1] = 256 << 16;         // tensor_dim0 = 256
    g1[2] = 5120 << 16;        // tensor_dim0 hi=0 | tensor_dim1 = 5120
    g1[3] = 256 << 16;         // tensor_dim1 hi=0 | tile_dim0 = 256
    g1[4] = 32;                // tile_dim1 = 32 | tile_dim2 = 0
    g1[5] = 256;               // tensor_dim0_stride = 256
    g1[6] = 0;
    g1[7] = 0;
    v4i z4 = {0, 0, 0, 0};
    v8i z8 = {0, 0, 0, 0, 0, 0, 0, 0};
    __builtin_amdgcn_tensor_load_to_lds(g0, g1, z4, z4, z8, 0);
}

// ---------------------------------------------------------------- main kernel
// grid: (N/128, 10); block: 256 (8 waves). Wave w owns rows rb + w*16.
__global__ void __launch_bounds__(256)
svc_main(const __bf16* __restrict__ xb, const __bf16* __restrict__ svb,
         const __bf16* __restrict__ ab, const float* __restrict__ gxn,
         const float* __restrict__ gsvn, float* __restrict__ T) {
    __shared__ __attribute__((aligned(128))) char svt[2][32 * ROWB]; // double-buffered sv tile
    __shared__ __attribute__((aligned(16)))  __bf16 ldsk[8][16 * 40]; // per-wave k transpose tile

    const int wave = threadIdx.x >> 5;
    const int lane = threadIdx.x & 31;
    const int lcol = lane & 15;           // column / M index within 16
    const int hi   = lane >> 4;           // lane half selects K/M sub-block
    const int rw   = blockIdx.x * 128 + wave * 16;   // wave row base
    const int cls  = blockIdx.y;

    // Resident A fragments: x tile 16 rows x 256 feats (A-layout: lane M = lcol,
    // per-lane 16 contiguous feats at k0+hi*16).
    v16bf afrag[8];
    {
        const __bf16* xrow = xb + (size_t)(rw + lcol) * FEAT + hi * 16;
#pragma unroll
        for (int kk = 0; kk < 8; ++kk) {
            v8bf g0 = *(const v8bf*)(xrow + kk * 32);
            v8bf g1 = *(const v8bf*)(xrow + kk * 32 + 8);
            afrag[kk] = __builtin_shufflevector(g0, g1, 0,1,2,3,4,5,6,7,8,9,10,11,12,13,14,15);
        }
    }
    // gamma*||x||^2 per accumulator row (C-layout rows: v + 8*hi)
    v8f xnf = *(const v8f*)(gxn + rw + hi * 8);

    v8f tacc = {0.f, 0.f, 0.f, 0.f, 0.f, 0.f, 0.f, 0.f};
    __bf16* myk = &ldsk[wave][0];

    // Prologue: fill buffer 0 with chunk 0 (wave 0 only; TDM ignores EXEC).
    if (wave == 0)
        svc_tdm_issue(svb, cls * NPAD, (unsigned)(uintptr_t)&svt[0][0]);

    for (int ch = 0; ch < NPAD / 32; ++ch) {
        const int sbase = cls * NPAD + ch * 32;
        // Current chunk's DMA complete (NOP for waves 1..7: their TENSORcnt==0).
        __builtin_amdgcn_s_wait_tensorcnt(0);
        __syncthreads();   // publish buf[ch&1]; buf[(ch+1)&1] reads (iter ch-1) retired
        if (wave == 0 && ch + 1 < NPAD / 32)
            svc_tdm_issue(svb, sbase + 32, (unsigned)(uintptr_t)&svt[(ch + 1) & 1][0]);

        const char* buf = &svt[ch & 1][0];
#pragma unroll
        for (int t = 0; t < 2; ++t) {
            // B fragment from LDS: sv^T 32x16 (lane N = sv index, per-lane 16 feats)
            const char* srow = buf + (t * 16 + lcol) * ROWB + hi * 32;
            v8f acc = {0.f, 0.f, 0.f, 0.f, 0.f, 0.f, 0.f, 0.f};
#pragma unroll
            for (int kk = 0; kk < 8; ++kk) {
                v16bf bfrag = *(const v16bf*)(srow + kk * 64);
                acc = __builtin_amdgcn_wmma_f32_16x16x32_bf16(
                    false, afrag[kk], false, bfrag, (short)0, acc, false, false);
            }
            // RBF: k = exp(2g*dot - g|x|^2 - g|sv|^2); write bf16 tile to LDS
            float svn = gsvn[sbase + t * 16 + lcol];
#pragma unroll
            for (int v = 0; v < 8; ++v) {
                float arg = 2.0f * SVC_GAMMA * acc[v] - xnf[v] - svn;
                myk[(v + hi * 8) * 40 + t * 16 + lcol] = (__bf16)__expf(arg);
            }
        }
        // Transpose k tile C-layout -> A-layout via LDS (wave-private)
        const __bf16* krow = myk + lcol * 40 + hi * 16;
        v8bf k0 = *(const v8bf*)(krow);
        v8bf k1 = *(const v8bf*)(krow + 8);
        v16bf a2 = __builtin_shufflevector(k0, k1, 0,1,2,3,4,5,6,7,8,9,10,11,12,13,14,15);
        // B fragment: a^T 32x16 (lane N = r, per-lane 16 contiguous SV cols)
        v16bf b2 = *(const v16bf*)(ab + (size_t)lcol * SVPAD + sbase + hi * 16);
        tacc = __builtin_amdgcn_wmma_f32_16x16x32_bf16(
            false, a2, false, b2, (short)0, tacc, false, false);

        __syncthreads();   // all waves done reading buf[ch&1] before it is refilled
    }

    // T stored as [row][class][r16] f32
#pragma unroll
    for (int v = 0; v < 8; ++v) {
        int row = rw + v + hi * 8;
        T[(size_t)row * (NCLS * 16) + cls * 16 + lcol] = tacc[v];
    }
}

// ---------------------------------------------------------------- voting kernel
__global__ void svc_vote(const float* __restrict__ T, const float* __restrict__ b,
                         int* __restrict__ out, int N) {
    int n = blockIdx.x * blockDim.x + threadIdx.x;
    if (n >= N) return;
    const float* tn = T + (size_t)n * (NCLS * 16);
    int cnt[NCLS];
#pragma unroll
    for (int c = 0; c < NCLS; ++c) cnt[c] = 0;
    int p = 0;
#pragma unroll
    for (int i = 0; i < NCLS; ++i) {
#pragma unroll
        for (int j = i + 1; j < NCLS; ++j) {
            // T[n, i, j] + T[n, j-1, i] + b[p]   (storage [row][c][r])
            float dec = tn[j * 16 + i] + tn[i * 16 + (j - 1)] + b[p];
            cnt[dec > 0.f ? i : j]++;
            ++p;
        }
    }
    int best = 0, bc = cnt[0];
#pragma unroll
    for (int c = 1; c < NCLS; ++c) if (cnt[c] > bc) { bc = cnt[c]; best = c; }
    out[n] = best;
    out[N + n] = best;
}

// ---------------------------------------------------------------- launch
extern "C" void kernel_launch(void* const* d_in, const int* in_sizes, int n_in,
                              void* d_out, int out_size, void* d_ws, size_t ws_size,
                              hipStream_t stream) {
    const float* x  = (const float*)d_in[0];
    const float* sv = (const float*)d_in[1];
    const float* a  = (const float*)d_in[2];
    const float* b  = (const float*)d_in[3];
    const int N = in_sizes[0] / FEAT;          // 8192

    // workspace layout (bytes)
    char* ws = (char*)d_ws;
    __bf16* xb   = (__bf16*)(ws);                                   // 4,194,304
    __bf16* svb  = (__bf16*)(ws + 4194304);                         // 2,621,440
    __bf16* ab   = (__bf16*)(ws + 6815744);                         //   163,840
    float*  gxn  = (float*) (ws + 6979584);                         //    32,768
    float*  gsvn = (float*) (ws + 7012352);                         //    20,480
    float*  T    = (float*) (ws + 7032832);                         // 5,242,880 -> ~12.3 MB total

    svc_prep_x <<<(N + 255) / 256, 256, 0, stream>>>(x, xb, gxn, N);
    svc_prep_sv<<<(SVPAD + 255) / 256, 256, 0, stream>>>(sv, svb, gsvn);
    svc_prep_a <<<(16 * SVPAD + 255) / 256, 256, 0, stream>>>(a, ab);

    dim3 grid(N / 128, NCLS);
    svc_main<<<grid, 256, 0, stream>>>(xb, svb, ab, gxn, gsvn, T);

    svc_vote<<<(N + 255) / 256, 256, 0, stream>>>(T, b, (int*)d_out, N);
}
